// SNN_Delay_22780506538285
// MI455X (gfx1250) — matrix operand order repaired and checked by reading, not compiled, not run on hardware
//
#include <hip/hip_runtime.h>
#include <hip/hip_bf16.h>
#include <math.h>

typedef __attribute__((ext_vector_type(16))) _Float16 v16h;
typedef __attribute__((ext_vector_type(8)))  float    v8f;

union Frag16 { v16h v; uint4 q[2]; };

constexpr int Bsz  = 1024;
constexpr int T    = 25;
constexpr int L    = 35;    // T + left pad
constexpr int LP   = 10;
constexpr int MAXD = 11;
constexpr int M    = Bsz * T;   // 25600 GEMM rows

// CDNA5 async global->LDS path (gfx1250). Guarded so host pass / older
// toolchains fall back to the VGPR-staged ds_store path.
#if defined(__gfx1250__) && \
    __has_builtin(__builtin_amdgcn_global_load_async_to_lds_b128) && \
    __has_builtin(__builtin_amdgcn_s_wait_asynccnt)
#define USE_ASYNC_LDS 1
// Builtin signature uses GCC-vector "V4i" pointees in AS(1)/AS(3).
typedef int v4i __attribute__((vector_size(16)));
typedef __attribute__((address_space(1))) v4i  gv4i;
typedef __attribute__((address_space(3))) v4i  lv4i;
typedef __attribute__((address_space(1))) void gvoid;
typedef __attribute__((address_space(3))) void lvoid;
#else
#define USE_ASYNC_LDS 0
#endif

// ---------------------------------------------------------------------------
// Build dense DCLS kernel, f16, layout (Npad, MAXD, ICP)  (k = d*ICP + i)
// ---------------------------------------------------------------------------
__global__ void build_dcls_kernel(const float* __restrict__ W,
                                  const float* __restrict__ P,
                                  const float* __restrict__ SIG,
                                  _Float16* __restrict__ Kern,
                                  int O, int Npad, int I, int ICP)
{
    int idx = blockIdx.x * blockDim.x + threadIdx.x;
    int total = Npad * ICP;
    if (idx >= total) return;
    int o = idx / ICP, i = idx - o * ICP;

    float vals[MAXD];
    if (o < O && i < I) {
        float Pc = P[o * I + i] + (float)(MAXD / 2);
        float S  = fabsf(SIG[o * I + i]) + 0.27f;
        float w  = W[o * I + i];
        float sum = 0.f;
#pragma unroll
        for (int d = 0; d < MAXD; ++d) {
            float z = ((float)d - Pc) / S;
            float e = __expf(-0.5f * z * z);
            vals[d] = e; sum += e;
        }
        float inv = w / (sum + 1e-7f);
#pragma unroll
        for (int d = 0; d < MAXD; ++d) vals[d] *= inv;
    } else {
#pragma unroll
        for (int d = 0; d < MAXD; ++d) vals[d] = 0.f;
    }
#pragma unroll
    for (int d = 0; d < MAXD; ++d)
        Kern[(size_t)(o * MAXD + d) * ICP + i] = (_Float16)vals[d];
}

// ---------------------------------------------------------------------------
// Layer-1 input: (B,T,140) f32 -> padded f16 (B, 35, 192), zeros in pads
// ---------------------------------------------------------------------------
__global__ void prep_input1(const float* __restrict__ data,
                            _Float16* __restrict__ Xp)
{
    constexpr int ICP = 192, I = 140;
    int idx = blockIdx.x * blockDim.x + threadIdx.x;   // b*L*ICP + l*ICP + i
    int i  = idx % ICP;
    int bl = idx / ICP;
    int l  = bl % L;
    int b  = bl / L;
    float v = 0.f;
    if (l >= LP && i < I) v = data[(size_t)(b * T + (l - LP)) * I + i];
    Xp[idx] = (_Float16)v;
}

// ---------------------------------------------------------------------------
// im2col GEMM via WMMA f16->f32.
//   A: Xp (B, 35, ICP) f16, row m=b*T+t reads Xp[b, t+d, i] for k=d*ICP+i
//   B: Kern (Npad, MAXD*ICP) f16 row-major
//   Y: (M, Npad) f32
// Block: 256 thr = 8 waves, tile 32(M) x 128(N), K-chunk 64.
// A tile is double-buffered in LDS via async global->LDS copies (ASYNCcnt),
// overlapping chunk c+1's DMA with chunk c's WMMAs.
// ---------------------------------------------------------------------------
template <int ICP>
__global__ __launch_bounds__(256)
void gemm_im2col_wmma(const _Float16* __restrict__ Xp,
                      const _Float16* __restrict__ Kern,
                      float* __restrict__ Y, int Npad)
{
    constexpr int KT = MAXD * ICP;           // 2112 or 2816, multiple of 64
    constexpr int NC = KT / 64;              // K chunks
    __shared__ __align__(16) _Float16 As[2][32 * 64];

    const int tid  = threadIdx.x;
    const int r    = tid >> 3, seg = tid & 7;          // A-tile fill mapping
    const int m    = blockIdx.x * 32 + r;
    const int b    = m / T;
    const int t    = m - b * T;

    const int wave = tid >> 5, lane = tid & 31;
    const int wm   = wave & 1;                 // 16-row half
    const int wn   = wave >> 1;                // 32-col quarter
    const int l16  = lane & 15;
    const int hi   = lane >> 4;                // lane >= 16 ?

    const int nbase = blockIdx.y * 128 + wn * 32;

    v8f acc0 = {}; v8f acc1 = {};

    // B fragment bases: lane supplies col n, K 0..7 and 16..23 (+8 for hi lanes)
    const _Float16* bbase0 = Kern + (size_t)(nbase + l16) * KT + hi * 8;
    const _Float16* bbase1 = bbase0 + (size_t)16 * KT;

    auto src_addr = [&](int c) -> const _Float16* {
        const int k0 = c * 64;
        const int d  = k0 / ICP;             // chunk lies inside one tap d
        const int i0 = k0 - d * ICP;
        return Xp + ((size_t)(b * L + t + d) * ICP + i0 + seg * 8);
    };

#if USE_ASYNC_LDS
    auto stage = [&](int c) {
        gv4i* gsrc = (gv4i*)((gvoid*)(void*)src_addr(c));
        lv4i* ldst = (lv4i*)((lvoid*)(void*)(&As[c & 1][r * 64 + seg * 8]));
        __builtin_amdgcn_global_load_async_to_lds_b128(gsrc, ldst, 0, 0);
    };
    stage(0);
#endif

    for (int c = 0; c < NC; ++c) {
        const int k0 = c * 64;

#if USE_ASYNC_LDS
        __syncthreads();                     // buf[(c+1)&1] free to overwrite
        if (c + 1 < NC) {
            stage(c + 1);                    // DMA next chunk while computing
            __builtin_amdgcn_s_wait_asynccnt(1);   // chunk c landed (mine)
        } else {
            __builtin_amdgcn_s_wait_asynccnt(0);
        }
        __syncthreads();                     // chunk c landed (everyone's)
        const _Float16* arow = &As[c & 1][0] + (wm * 16 + l16) * 64 + hi * 8;
#else
        uint4 srcv = *(const uint4*)src_addr(c);
        __syncthreads();                     // prior chunk's reads complete
        *(uint4*)(&As[0][r * 64 + seg * 8]) = srcv;
        __syncthreads();                     // tile visible
        const _Float16* arow = &As[0][0] + (wm * 16 + l16) * 64 + hi * 8;
#endif

        if (k0 + 64 < KT) {                  // warm L0/L2 for next B chunk
            __builtin_prefetch((const void*)(bbase0 + k0 + 64), 0, 1);
            __builtin_prefetch((const void*)(bbase1 + k0 + 64), 0, 1);
        }

#pragma unroll
        for (int ks = 0; ks < 64; ks += 32) {
            Frag16 a;
            const _Float16* ap = arow + ks;
            a.q[0] = *(const uint4*)ap;            // K 0..7   (+8 for hi lanes)
            a.q[1] = *(const uint4*)(ap + 16);     // K 16..23 (+8 for hi lanes)

            Frag16 b0, b1;
            const _Float16* bp0 = bbase0 + k0 + ks;
            b0.q[0] = *(const uint4*)bp0;
            b0.q[1] = *(const uint4*)(bp0 + 16);
            const _Float16* bp1 = bbase1 + k0 + ks;
            b1.q[0] = *(const uint4*)bp1;
            b1.q[1] = *(const uint4*)(bp1 + 16);

            acc0 = __builtin_amdgcn_wmma_f32_16x16x32_f16(
                       false, a.v, false, b0.v, (short)0, acc0, false, false);
            acc1 = __builtin_amdgcn_wmma_f32_16x16x32_f16(
                       false, a.v, false, b1.v, (short)0, acc1, false, false);
        }
    }

    const int mg = blockIdx.x * 32 + wm * 16 + hi * 8;   // C layout: VGPR j -> row j (+8 hi)
#pragma unroll
    for (int j = 0; j < 8; ++j) {
        float* yp = Y + (size_t)(mg + j) * Npad + nbase + l16;
        yp[0]  = acc0[j];
        yp[16] = acc1[j];
    }
}

// ---------------------------------------------------------------------------
// Hidden-layer LIF: fused bias+BN affine, 25-step scan, write f16 spikes into
// next layer's padded input (B, 35, 256), pads zeroed here.
// ---------------------------------------------------------------------------
__global__ void lif_hidden(const float* __restrict__ Y, int Npad,
                           const float* __restrict__ gam, const float* __restrict__ bet,
                           const float* __restrict__ mu,  const float* __restrict__ var,
                           const float* __restrict__ bias,
                           _Float16* __restrict__ XpN)
{
    int idx = blockIdx.x * blockDim.x + threadIdx.x;   // b*256 + c
    if (idx >= Bsz * 256) return;
    int c = idx & 255, b = idx >> 8;

    float sc = gam[c] * rsqrtf(var[c] + 1e-5f);
    float sh = (bias[c] - mu[c]) * sc + bet[c];

#pragma unroll
    for (int l = 0; l < LP; ++l)
        XpN[(size_t)(b * L + l) * 256 + c] = (_Float16)0.f;

    float mem = 0.f;
#pragma unroll
    for (int tt = 0; tt < T; ++tt) {
        float x = Y[(size_t)(b * T + tt) * Npad + c] * sc + sh;
        float reset = (mem > 1.0f) ? 1.0f : 0.0f;      // detached heaviside(mem-1)
        mem = 0.95f * mem + x - reset;
        float s = (mem > 1.0f) ? 1.0f : 0.0f;
        XpN[(size_t)(b * L + LP + tt) * 256 + c] = (_Float16)s;
    }
}

// ---------------------------------------------------------------------------
// Output LIF: bias only (no BN), emit spk3 (T,B,20) then mem3 (T,B,20), f32
// ---------------------------------------------------------------------------
__global__ void lif_out(const float* __restrict__ Y, const float* __restrict__ bias,
                        float* __restrict__ outp)
{
    int idx = blockIdx.x * blockDim.x + threadIdx.x;   // b*20 + c
    if (idx >= Bsz * 20) return;
    int c = idx % 20, b = idx / 20;

    float sh = bias[c];
    float mem = 0.f;
    float* spk = outp;
    float* memo = outp + (size_t)T * Bsz * 20;
#pragma unroll
    for (int tt = 0; tt < T; ++tt) {
        float x = Y[(size_t)(b * T + tt) * 128 + c] + sh;
        float reset = (mem > 1.0f) ? 1.0f : 0.0f;
        mem = 0.95f * mem + x - reset;
        float s = (mem > 1.0f) ? 1.0f : 0.0f;
        size_t o = (size_t)tt * Bsz * 20 + (size_t)b * 20 + c;
        spk[o]  = s;
        memo[o] = mem;
    }
}

// ---------------------------------------------------------------------------
extern "C" void kernel_launch(void* const* d_in, const int* in_sizes, int n_in,
                              void* d_out, int out_size, void* d_ws, size_t ws_size,
                              hipStream_t stream)
{
    const float* data = (const float*)d_in[0];
    const float* W1   = (const float*)d_in[1];
    const float* b1   = (const float*)d_in[2];
    const float* P1   = (const float*)d_in[3];
    const float* SIG1 = (const float*)d_in[4];
    const float* g1   = (const float*)d_in[5];
    const float* be1  = (const float*)d_in[6];
    const float* mu1  = (const float*)d_in[7];
    const float* v1   = (const float*)d_in[8];
    const float* W2   = (const float*)d_in[9];
    const float* b2   = (const float*)d_in[10];
    const float* P2   = (const float*)d_in[11];
    const float* SIG2 = (const float*)d_in[12];
    const float* g2   = (const float*)d_in[13];
    const float* be2  = (const float*)d_in[14];
    const float* mu2  = (const float*)d_in[15];
    const float* v2   = (const float*)d_in[16];
    const float* W3   = (const float*)d_in[17];
    const float* b3   = (const float*)d_in[18];
    const float* P3   = (const float*)d_in[19];
    const float* SIG3 = (const float*)d_in[20];

    // ---- workspace carve (256-B aligned) ----
    char* ws = (char*)d_ws;
    size_t off = 0;
    auto carve = [&](size_t bytes) -> void* {
        void* p = ws + off;
        off = (off + bytes + 255) & ~(size_t)255;
        return p;
    };
    _Float16* Xp1 = (_Float16*)carve((size_t)Bsz * L * 192 * 2);   // 13.8 MB
    _Float16* Xp2 = (_Float16*)carve((size_t)Bsz * L * 256 * 2);   // 18.4 MB
    _Float16* Xp3 = (_Float16*)carve((size_t)Bsz * L * 256 * 2);   // 18.4 MB
    _Float16* K1  = (_Float16*)carve((size_t)256 * MAXD * 192 * 2);
    _Float16* K2  = (_Float16*)carve((size_t)256 * MAXD * 256 * 2);
    _Float16* K3  = (_Float16*)carve((size_t)128 * MAXD * 256 * 2);
    float*    Yb  = (float*)carve((size_t)M * 256 * 4);            // 26.2 MB (layers 1 & 2)
    float*    Y3  = (float*)carve((size_t)M * 128 * 4);            // 13.1 MB
    (void)ws_size; (void)in_sizes; (void)n_in; (void)out_size;

    // ---- build DCLS kernels (f16, (Npad, d, ICP)) ----
    build_dcls_kernel<<<(256 * 192 + 255) / 256, 256, 0, stream>>>(W1, P1, SIG1, K1, 256, 256, 140, 192);
    build_dcls_kernel<<<(256 * 256 + 255) / 256, 256, 0, stream>>>(W2, P2, SIG2, K2, 256, 256, 256, 256);
    build_dcls_kernel<<<(128 * 256 + 255) / 256, 256, 0, stream>>>(W3, P3, SIG3, K3,  20, 128, 256, 256);

    // ---- layer 1 ----
    prep_input1<<<(Bsz * L * 192) / 256, 256, 0, stream>>>(data, Xp1);
    gemm_im2col_wmma<192><<<dim3(M / 32, 2), 256, 0, stream>>>(Xp1, K1, Yb, 256);
    lif_hidden<<<(Bsz * 256) / 256, 256, 0, stream>>>(Yb, 256, g1, be1, mu1, v1, b1, Xp2);

    // ---- layer 2 ----
    gemm_im2col_wmma<256><<<dim3(M / 32, 2), 256, 0, stream>>>(Xp2, K2, Yb, 256);
    lif_hidden<<<(Bsz * 256) / 256, 256, 0, stream>>>(Yb, 256, g2, be2, mu2, v2, b2, Xp3);

    // ---- layer 3 + outputs ----
    gemm_im2col_wmma<256><<<dim3(M / 32, 1), 256, 0, stream>>>(Xp3, K3, Y3, 128);
    lif_out<<<(Bsz * 20 + 255) / 256, 256, 0, stream>>>(Y3, b3, (float*)d_out);
}
